// PieceRoutedAttention_87591563034847
// MI455X (gfx1250) — matrix operand order repaired and checked by reading, not compiled, not run on hardware
//
#include <hip/hip_runtime.h>

// ---------------------------------------------------------------------------
// CDNA5 (gfx1250) bf16 WMMA implementation of the chess "piece-routed"
// attention block.  All GEMMs use v_wmma_f32_16x16x32_bf16 (wave32).
// Round 2: L2-friendly grid order (head index fastest so x / y tiles are
// reused from L2, not HBM) + double-buffered LDS staging in both GEMM
// kernels, grouped B-fragment loads per WMMA triple.
// ---------------------------------------------------------------------------

typedef __bf16        v16bf __attribute__((ext_vector_type(16)));
typedef float         v8f   __attribute__((ext_vector_type(8)));
typedef float         f32x4 __attribute__((ext_vector_type(4)));
typedef unsigned int  u32x4 __attribute__((ext_vector_type(4)));

union Frag16 { v16bf v; u32x4 q[2]; };

// A-matrix fragment (16x32 bf16), row-major source: lane l holds row M=l&15.
// Layout: lanes 0-15 hold K 0..15, lanes 16-31 hold K 16..31 (contiguous,
//         matching load_b_frag).
__device__ __forceinline__ v16bf load_a_frag(const __bf16* base, int ld,
                                             int kbase, int lane) {
  int m  = lane & 15;
  int hi = lane >> 4;
  const __bf16* p = base + m * ld + kbase + hi * 16;
  Frag16 f;
  f.q[0] = *(const u32x4*)(p);        // 8 contiguous bf16
  f.q[1] = *(const u32x4*)(p + 8);    // 8 contiguous bf16
  return f.v;
}

// B-matrix fragment (32x16 bf16) from a TRANSPOSED tile Bt[n][k]:
// lane l holds column N=l&15; lanes 0-15 hold K 0..15, lanes 16-31 K 16..31,
// i.e. 16 contiguous bf16 per lane.
__device__ __forceinline__ v16bf load_b_frag(const __bf16* baseT, int ld,
                                             int kbase, int lane) {
  int n  = lane & 15;
  int hi = lane >> 4;
  const __bf16* p = baseT + n * ld + kbase + hi * 16;
  Frag16 f;
  f.q[0] = *(const u32x4*)(p);
  f.q[1] = *(const u32x4*)(p + 8);
  return f.v;
}

__device__ __forceinline__ v8f wmma_bf16(v16bf a, v16bf b, v8f c) {
  return __builtin_amdgcn_wmma_f32_16x16x32_bf16(
      /*neg_a=*/false, a, /*neg_b=*/false, b,
      /*c_mod=*/(short)0, c, /*reuse_a=*/false, /*reuse_b=*/false);
}

// ---------------------------------------------------------------------------
// Kernel 1: free-head bias = log(sigmoid(free_logits) + 1e-6)   (3*64*64)
// ---------------------------------------------------------------------------
__global__ void fbias_kernel(const float* __restrict__ fl,
                             float* __restrict__ fb) {
  int i = blockIdx.x * 256 + threadIdx.x;
  if (i < 3 * 64 * 64) {
    float s = 1.0f / (1.0f + __expf(-fl[i]));
    fb[i] = __logf(s + 1e-6f);
  }
}

// ---------------------------------------------------------------------------
// LDS staging helpers (f32 global -> bf16 LDS)
// ---------------------------------------------------------------------------
__device__ __forceinline__ void stage_a(const float* __restrict__ x,
                                        __bf16* As, int row0, int k0,
                                        int tid) {
  // x[row0 .. +64][k0 .. +32] f32 -> bf16 row-major (ld = 32)
  for (int e = tid; e < 512; e += 128) {
    int r = e >> 3, cc = (e & 7) << 2;
    f32x4 xv = *(const f32x4*)(x + (size_t)(row0 + r) * 768 + k0 + cc);
    __bf16* dst = As + r * 32 + cc;
    dst[0] = (__bf16)xv.x; dst[1] = (__bf16)xv.y;
    dst[2] = (__bf16)xv.z; dst[3] = (__bf16)xv.w;
  }
}

__device__ __forceinline__ void stage_bT(const float* __restrict__ W,
                                         __bf16* bt, int k0, int h, int tid) {
  // W rows k0..k0+31, cols h*64..h*64+63  ->  bt[n][k] (64 x 32, transposed)
  for (int e = tid; e < 512; e += 128) {
    int kk = e >> 4, nn = (e & 15) << 2;
    f32x4 wv = *(const f32x4*)(W + (size_t)(k0 + kk) * 768 + h * 64 + nn);
    bt[(nn + 0) * 32 + kk] = (__bf16)wv.x;
    bt[(nn + 1) * 32 + kk] = (__bf16)wv.y;
    bt[(nn + 2) * 32 + kk] = (__bf16)wv.z;
    bt[(nn + 3) * 32 + kk] = (__bf16)wv.w;
  }
}

// ---------------------------------------------------------------------------
// Kernel 2: fused q/k/v projection + RMSNorm(q,k).
// Grid: (12 heads, 1024 m-tiles)  -- head index fastest so consecutive
// blocks share the same 192KB x tile through L2.  Block: 128 threads.
// Double-buffered LDS staging: one barrier per K-chunk.
// ---------------------------------------------------------------------------
__device__ __forceinline__ void rmsnorm_strip(v8f (&acc)[4]) {
#pragma unroll
  for (int i = 0; i < 8; ++i) {
    float ss = 0.f;
#pragma unroll
    for (int t = 0; t < 4; ++t) { float v = acc[t][i]; ss += v * v; }
#pragma unroll
    for (int m = 1; m < 16; m <<= 1) ss += __shfl_xor(ss, m, 32);
    float r = rsqrtf(ss * (1.0f / 64.0f) + 1.1920929e-7f);
#pragma unroll
    for (int t = 0; t < 4; ++t) acc[t][i] *= r;
  }
}

__global__ __launch_bounds__(128) void qkv_kernel(
    const float* __restrict__ x,
    const float* __restrict__ Wq, const float* __restrict__ Wk,
    const float* __restrict__ Wv,
    __bf16* __restrict__ qws, __bf16* __restrict__ kws,
    __bf16* __restrict__ vws) {
  __shared__ __bf16 As[2][64 * 32];
  __shared__ __bf16 Bqs[2][64 * 32];
  __shared__ __bf16 Bks[2][64 * 32];
  __shared__ __bf16 Bvs[2][64 * 32];

  int h  = blockIdx.x;           // 12 heads (fastest -> L2 reuse of x tile)
  int bm = blockIdx.y;           // 1024 row tiles (== batch index, T=64)
  int tid = threadIdx.x, wave = tid >> 5, lane = tid & 31;

  v8f zero = {0.f, 0.f, 0.f, 0.f, 0.f, 0.f, 0.f, 0.f};
  v8f accq[4], acck[4], accv[4];
#pragma unroll
  for (int t = 0; t < 4; ++t) { accq[t] = zero; acck[t] = zero; accv[t] = zero; }

  // prologue: stage chunk 0 into buffer 0
  stage_a(x, As[0], bm * 64, 0, tid);
  stage_bT(Wq, Bqs[0], 0, h, tid);
  stage_bT(Wk, Bks[0], 0, h, tid);
  stage_bT(Wv, Bvs[0], 0, h, tid);

  for (int i = 0; i < 24; ++i) {   // 768 / 32 chunks
    int cur = i & 1;
    __syncthreads();               // buffer[cur] ready; buffer[cur^1] free
    if (i + 1 < 24) {
      int k1 = (i + 1) * 32;
      stage_a(x, As[cur ^ 1], bm * 64, k1, tid);
      stage_bT(Wq, Bqs[cur ^ 1], k1, h, tid);
      stage_bT(Wk, Bks[cur ^ 1], k1, h, tid);
      stage_bT(Wv, Bvs[cur ^ 1], k1, h, tid);
    }
    v16bf a = load_a_frag(As[cur] + wave * 16 * 32, 32, 0, lane);
#pragma unroll
    for (int t = 0; t < 4; ++t) {
      v16bf bq = load_b_frag(Bqs[cur] + t * 16 * 32, 32, 0, lane);
      v16bf bk = load_b_frag(Bks[cur] + t * 16 * 32, 32, 0, lane);
      v16bf bv = load_b_frag(Bvs[cur] + t * 16 * 32, 32, 0, lane);
      accq[t] = wmma_bf16(a, bq, accq[t]);
      acck[t] = wmma_bf16(a, bk, acck[t]);
      accv[t] = wmma_bf16(a, bv, accv[t]);
    }
  }

  rmsnorm_strip(accq);
  rmsnorm_strip(acck);

  // store as (B, H, 64, 64) bf16
#pragma unroll
  for (int i = 0; i < 8; ++i) {
    int row = wave * 16 + i + (lane >> 4) * 8;
    size_t base = ((size_t)(bm * 12 + h) * 64 + row) * 64;
#pragma unroll
    for (int t = 0; t < 4; ++t) {
      int col = t * 16 + (lane & 15);
      qws[base + col] = (__bf16)accq[t][i];
      kws[base + col] = (__bf16)acck[t][i];
      vws[base + col] = (__bf16)accv[t][i];
    }
  }
}

// ---------------------------------------------------------------------------
// Kernel 3: attention per (b, h).  Block: 128 threads (4 waves), 12288 blocks.
// ---------------------------------------------------------------------------
__device__ __forceinline__ float attn_bias(int h, int row, int col,
                                           const float* __restrict__ fb) {
  if (h >= 9) return fb[((h - 9) * 64 + row) * 64 + col];
  if (row == col) return 0.f;   // eye always allowed
  int rq = row >> 3, cq = row & 7;
  int rk = col >> 3, ck = col & 7;
  int dr = rk - rq, dc = ck - cq;
  int adr = dr < 0 ? -dr : dr, adc = dc < 0 ? -dc : dc;
  bool ok;
  if (h < 2)       ok = (adr == 1 && adc == 2) || (adr == 2 && adc == 1); // knight
  else if (h < 4)  ok = (adr == adc) && (adr != 0);                        // bishop
  else if (h < 6)  ok = (adr == 0) != (adc == 0);                          // rook
  else if (h == 6) ok = ((adr == adc) && adr) || ((adr == 0) != (adc == 0)); // queen
  else if (h == 7) ok = ((adr > adc ? adr : adc) == 1);                    // king
  else             ok = (dr == 1) && (adc <= 1);                           // pawn
  return ok ? 0.f : -1e30f;
}

__global__ __launch_bounds__(128) void attn_kernel(
    const __bf16* __restrict__ q, const __bf16* __restrict__ k,
    const __bf16* __restrict__ v, const float* __restrict__ fb,
    __bf16* __restrict__ y) {
  __shared__ __bf16 Qs[64 * 64];
  __shared__ __bf16 Ks[64 * 64];   // row-major over key-square == Bt for QK^T
  __shared__ __bf16 VTs[64 * 64];  // V transposed: [feature][key]
  __shared__ float  Ss[64 * 64];
  __shared__ __bf16 Ps[64 * 64];

  int bh = blockIdx.x;
  int h = bh % 12, b = bh / 12;
  size_t base = (size_t)bh * 4096;
  int tid = threadIdx.x, wave = tid >> 5, lane = tid & 31;

  // cooperative loads (bf16, 8 KB each)
  for (int e = tid; e < 512; e += 128) {
    ((u32x4*)Qs)[e] = ((const u32x4*)(q + base))[e];
    ((u32x4*)Ks)[e] = ((const u32x4*)(k + base))[e];
  }
  for (int e = tid; e < 4096; e += 128) {
    int t = e >> 6, d = e & 63;
    VTs[d * 64 + t] = v[base + e];
  }
  __syncthreads();

  v8f zero = {0.f, 0.f, 0.f, 0.f, 0.f, 0.f, 0.f, 0.f};
  v8f acc[4];
#pragma unroll
  for (int t = 0; t < 4; ++t) acc[t] = zero;

  // S = Q K^T  (each wave: 16-row strip; hd=64 -> two K=32 chunks)
  {
    v16bf a0 = load_a_frag(Qs + wave * 16 * 64, 64, 0, lane);
    v16bf a1 = load_a_frag(Qs + wave * 16 * 64, 64, 32, lane);
#pragma unroll
    for (int t = 0; t < 4; ++t) {
      v16bf b0 = load_b_frag(Ks + t * 16 * 64, 64, 0, lane);
      v16bf b1 = load_b_frag(Ks + t * 16 * 64, 64, 32, lane);
      acc[t] = wmma_bf16(a0, b0, acc[t]);
      acc[t] = wmma_bf16(a1, b1, acc[t]);
    }
  }
  // dump scaled scores (1/sqrt(64) = 0.125) to LDS
#pragma unroll
  for (int i = 0; i < 8; ++i) {
    int row = wave * 16 + i + (lane >> 4) * 8;
#pragma unroll
    for (int t = 0; t < 4; ++t)
      Ss[row * 64 + t * 16 + (lane & 15)] = acc[t][i] * 0.125f;
  }
  __syncthreads();

  // softmax with bias: one thread per query row
  if (tid < 64) {
    int row = tid;
    float mx = -3.4e38f;
    for (int c = 0; c < 64; ++c) {
      float s = Ss[row * 64 + c] + attn_bias(h, row, c, fb);
      Ss[row * 64 + c] = s;
      mx = fmaxf(mx, s);
    }
    float sum = 0.f;
    for (int c = 0; c < 64; ++c) {
      float e = __expf(Ss[row * 64 + c] - mx);
      sum += e;
      Ss[row * 64 + c] = e;
    }
    float inv = 1.0f / sum;
    for (int c = 0; c < 64; ++c)
      Ps[row * 64 + c] = (__bf16)(Ss[row * 64 + c] * inv);
  }
  __syncthreads();

  // Y = P * V   (B operand = VTs, already [n=feature][k=key])
#pragma unroll
  for (int t = 0; t < 4; ++t) acc[t] = zero;
  {
    v16bf a0 = load_a_frag(Ps + wave * 16 * 64, 64, 0, lane);
    v16bf a1 = load_a_frag(Ps + wave * 16 * 64, 64, 32, lane);
#pragma unroll
    for (int t = 0; t < 4; ++t) {
      v16bf b0 = load_b_frag(VTs + t * 16 * 64, 64, 0, lane);
      v16bf b1 = load_b_frag(VTs + t * 16 * 64, 64, 32, lane);
      acc[t] = wmma_bf16(a0, b0, acc[t]);
      acc[t] = wmma_bf16(a1, b1, acc[t]);
    }
  }
  // store y as (B*T, 768) bf16: row = b*64+t, col = h*64+d
#pragma unroll
  for (int i = 0; i < 8; ++i) {
    int row = wave * 16 + i + (lane >> 4) * 8;
    size_t yb = ((size_t)b * 64 + row) * 768 + h * 64;
#pragma unroll
    for (int t = 0; t < 4; ++t)
      y[yb + t * 16 + (lane & 15)] = (__bf16)acc[t][i];
  }
}

// ---------------------------------------------------------------------------
// Kernel 4: out = y(bf16) @ Wo(f32->bf16), f32 output.
// Grid: (12 n-tiles, 1024 m-tiles) -- n fastest so the y tile is L2-reused.
// Double-buffered LDS staging.
// ---------------------------------------------------------------------------
__global__ __launch_bounds__(128) void outgemm_kernel(
    const __bf16* __restrict__ y, const float* __restrict__ Wo,
    float* __restrict__ out) {
  __shared__ __bf16 As[2][64 * 32];
  __shared__ __bf16 Bs[2][64 * 32];
  int bn = blockIdx.x, bm = blockIdx.y;
  int tid = threadIdx.x, wave = tid >> 5, lane = tid & 31;

  v8f zero = {0.f, 0.f, 0.f, 0.f, 0.f, 0.f, 0.f, 0.f};
  v8f acc[4];
#pragma unroll
  for (int t = 0; t < 4; ++t) acc[t] = zero;

  // A staging (bf16 source, straight copy)
  auto stage_ay = [&](__bf16* dst, int k0) {
    for (int e = tid; e < 256; e += 128) {
      int r = e >> 2, cc = (e & 3) << 3;
      *(u32x4*)(dst + r * 32 + cc) =
          *(const u32x4*)(y + (size_t)(bm * 64 + r) * 768 + k0 + cc);
    }
  };

  stage_ay(As[0], 0);
  stage_bT(Wo, Bs[0], 0, bn, tid);

  for (int i = 0; i < 24; ++i) {
    int cur = i & 1;
    __syncthreads();
    if (i + 1 < 24) {
      int k1 = (i + 1) * 32;
      stage_ay(As[cur ^ 1], k1);
      stage_bT(Wo, Bs[cur ^ 1], k1, bn, tid);
    }
    v16bf a = load_a_frag(As[cur] + wave * 16 * 32, 32, 0, lane);
#pragma unroll
    for (int t = 0; t < 4; ++t) {
      v16bf bb = load_b_frag(Bs[cur] + t * 16 * 32, 32, 0, lane);
      acc[t] = wmma_bf16(a, bb, acc[t]);
    }
  }
#pragma unroll
  for (int i = 0; i < 8; ++i) {
    int row = wave * 16 + i + (lane >> 4) * 8;
#pragma unroll
    for (int t = 0; t < 4; ++t)
      out[(size_t)(bm * 64 + row) * 768 + bn * 64 + t * 16 + (lane & 15)] =
          acc[t][i];
  }
}

// ---------------------------------------------------------------------------
extern "C" void kernel_launch(void* const* d_in, const int* in_sizes, int n_in,
                              void* d_out, int out_size, void* d_ws,
                              size_t ws_size, hipStream_t stream) {
  const float* x  = (const float*)d_in[0];
  const float* Wq = (const float*)d_in[1];
  const float* Wk = (const float*)d_in[2];
  const float* Wv = (const float*)d_in[3];
  const float* Wo = (const float*)d_in[4];
  const float* fl = (const float*)d_in[5];

  const size_t QKV_BYTES = (size_t)1024 * 12 * 64 * 64 * sizeof(__bf16); // 96 MiB
  char* ws = (char*)d_ws;
  __bf16* qws = (__bf16*)(ws);
  __bf16* kws = (__bf16*)(ws + QKV_BYTES);
  __bf16* vws = (__bf16*)(ws + 2 * QKV_BYTES);
  __bf16* yws = (__bf16*)(ws + 3 * QKV_BYTES);
  float*  fb  = (float*)(ws + 4 * QKV_BYTES);

  fbias_kernel<<<48, 256, 0, stream>>>(fl, fb);

  dim3 gq(12, 1024);   // head fastest -> x tile L2 reuse
  qkv_kernel<<<gq, 128, 0, stream>>>(x, Wq, Wk, Wv, qws, kws, vws);

  attn_kernel<<<1024 * 12, 128, 0, stream>>>(qws, kws, vws, fb, yws);

  dim3 go(12, 1024);   // n-tile fastest -> y tile L2 reuse
  outgemm_kernel<<<go, 128, 0, stream>>>(yws, Wo, (float*)d_out);
}
